// PaLM_Attention_37752762532341
// MI455X (gfx1250) — matrix operand order repaired and checked by reading, not compiled, not run on hardware
//
#include <hip/hip_runtime.h>

// ---------------------------------------------------------------------------
// PaLM attention (xPos RoPE, single-KV-head attention, out-proj) for gfx1250.
// Compute-bound -> all matmuls via v_wmma_f32_16x16x32_bf16.
// GEMM + flash K-tiles fed by the Tensor Data Mover (tensor_load_to_lds,
// TENSORcnt), double-buffered in LDS, with TDM padding recreating the
// bank-conflict-free padded row strides.
// ---------------------------------------------------------------------------

#define D_MODEL 2048
#define N_HEADS 16
#define HEAD_DIM 128
#define BATCH 2
#define SEQ 2048
#define MTOT (BATCH * SEQ) // 4096

typedef __attribute__((ext_vector_type(16))) __bf16 v16bf;
typedef __attribute__((ext_vector_type(8))) float v8f;
typedef __attribute__((ext_vector_type(4))) unsigned int u32x4;
typedef __attribute__((ext_vector_type(4))) int i32x4;
typedef __attribute__((ext_vector_type(8))) int i32x8;

union FragBF {
  v16bf bf;
  u32x4 q[2];
  unsigned short us[16];
};

__device__ __forceinline__ unsigned short f2bf(float f) {
  unsigned int u = __float_as_uint(f);
  unsigned int r = ((u >> 16) & 1u) + 0x7FFFu; // round-to-nearest-even
  return (unsigned short)((u + r) >> 16);
}

#if __has_builtin(__builtin_amdgcn_tensor_load_to_lds)
#define HAVE_TDM 1
#else
#define HAVE_TDM 0
#endif

#if HAVE_TDM
// One 2D TDM tile load: global row-major (strideElems between rows, 2-byte
// elems) -> LDS, inserting padAmount(+1) DWORDs every 2^(padInterval+1)
// DWORDs (recreates padded LDS row strides).
__device__ __forceinline__ void tdm_load_2d(unsigned lds_addr,
                                            const void* gptr, unsigned tile0,
                                            unsigned tile1, unsigned tdim0,
                                            unsigned tdim1,
                                            unsigned strideElems,
                                            unsigned padInterval,
                                            unsigned padAmount) {
  const unsigned long long ga = (unsigned long long)gptr;
  u32x4 g0;
  g0[0] = 1u;                                  // count=1, user descriptor
  g0[1] = lds_addr;                            // LDS byte address
  g0[2] = (unsigned)(ga & 0xFFFFFFFFu);        // global_addr[31:0]
  g0[3] = (unsigned)((ga >> 32) & 0x01FFFFFFu) // global_addr[56:32]
          | (2u << 30);                        // type = 2 ("image")
  i32x8 g1;
  g1[0] = (int)((1u << 16)                     // data_size = 2 bytes
                | (1u << 20)                   // pad_enable
                | (padInterval << 22) | (padAmount << 25));
  g1[1] = (int)((tdim0 & 0xFFFFu) << 16);      // tensor_dim0 lo
  g1[2] = (int)(((tdim0 >> 16) & 0xFFFFu) | ((tdim1 & 0xFFFFu) << 16));
  g1[3] = (int)(((tdim1 >> 16) & 0xFFFFu) | ((tile0 & 0xFFFFu) << 16));
  g1[4] = (int)(tile1 & 0xFFFFu);              // tile_dim1
  g1[5] = (int)strideElems;                    // dim0 stride lo
  g1[6] = 0;
  g1[7] = 0;
  i32x4 z4 = {0, 0, 0, 0};
#if defined(__clang_major__) && (__clang_major__ >= 23)
  i32x8 z8 = {0, 0, 0, 0, 0, 0, 0, 0};
  __builtin_amdgcn_tensor_load_to_lds(g0, g1, z4, z4, z8, 0);
#else
  __builtin_amdgcn_tensor_load_to_lds(g0, g1, z4, z4, 0);
#endif
}
#endif // HAVE_TDM

// ---------------------------------------------------------------------------
// Kernel 1: fp32 -> bf16 conversion (grid-stride)
// ---------------------------------------------------------------------------
__global__ void cvt_f32_bf16(const float* __restrict__ in,
                             unsigned short* __restrict__ out, int n) {
  int i = blockIdx.x * blockDim.x + threadIdx.x;
  int stride = gridDim.x * blockDim.x;
  for (; i < n; i += stride) out[i] = f2bf(in[i]);
}

// ---------------------------------------------------------------------------
// Kernel 2: C[M,N] (f32) = A[M,K](bf16) @ W[N,K](bf16)^T (+ bias[n])
// 128x128 tile, K-step 64 -> 16 WMMAs per wave per barrier. TDM
// double-buffered: wave 0 issues next tile, all compute current, wave 0
// waits TENSORcnt, block barriers, buffers swap. B-fragment loads are
// software-pipelined two-deep.
// ---------------------------------------------------------------------------
__global__ __launch_bounds__(256) void gemm_bf16_wmma(
    const unsigned short* __restrict__ A, const unsigned short* __restrict__ W,
    const float* __restrict__ bias, float* __restrict__ C, int M, int N,
    int K) {
  // Row = 64 elems = 32 DW -> pad_interval code 4; pad 4 DW (code 3)
  // -> LDS row stride 144 B == [72] ushorts.
  __shared__ unsigned short As[2][128][72];
  __shared__ unsigned short Bs[2][128][72];

  const int tid = threadIdx.x;
  const int waveS = __builtin_amdgcn_readfirstlane(tid) >> 5; // SGPR wave id
  const int lane = tid & 31;
  const int lhalf = lane >> 4;
  const int l16 = lane & 15;
  const int bm = blockIdx.x * 128;
  const int bn = blockIdx.y * 128;

  v8f acc[8];
#pragma unroll
  for (int t = 0; t < 8; ++t)
#pragma unroll
    for (int r = 0; r < 8; ++r) acc[t][r] = 0.0f;

  const int ka = lhalf * 8;   // A-frag K offset
  const int kbb = lhalf * 8;  // B-frag K offset

#if HAVE_TDM
  if (waveS == 0) {
    tdm_load_2d((unsigned)(size_t)&As[0][0][0], A + (size_t)bm * K, 64, 128,
                (unsigned)K, (unsigned)M, (unsigned)K, 4, 3);
    tdm_load_2d((unsigned)(size_t)&Bs[0][0][0], W + (size_t)bn * K, 64, 128,
                (unsigned)K, (unsigned)N, (unsigned)K, 4, 3);
    __builtin_amdgcn_s_wait_tensorcnt(0);
  }
  __syncthreads();

  int c = 0;
  for (int k0 = 0; k0 < K; k0 += 64) {
    const bool last = (k0 + 64 >= K);
    if (!last && waveS == 0) {
      tdm_load_2d((unsigned)(size_t)&As[c ^ 1][0][0],
                  A + (size_t)bm * K + k0 + 64, 64, 128, (unsigned)K,
                  (unsigned)M, (unsigned)K, 4, 3);
      tdm_load_2d((unsigned)(size_t)&Bs[c ^ 1][0][0],
                  W + (size_t)bn * K + k0 + 64, 64, 128, (unsigned)K,
                  (unsigned)N, (unsigned)K, 4, 3);
    }

    // A-fragments for both 32-wide K-steps of this 64-wide tile.
    const unsigned short* arow = &As[c][waveS * 16 + l16][0];
    FragBF af0, af1;
    af0.q[0] = *(const u32x4*)(arow + ka);
    af0.q[1] = *(const u32x4*)(arow + 16 + ka);
    af1.q[0] = *(const u32x4*)(arow + 32 + ka);
    af1.q[1] = *(const u32x4*)(arow + 48 + ka);

    // Two-deep pipelined B-fragment stream: 16 WMMAs back-to-back-ish.
    FragBF bcur, bnxt;
    {
      const unsigned short* brow = &Bs[c][l16][0];
      bcur.q[0] = *(const u32x4*)(brow + kbb);
      bcur.q[1] = *(const u32x4*)(brow + kbb + 16);
    }
#pragma unroll
    for (int idx = 0; idx < 16; ++idx) {
      const int t = idx >> 1;
      const int kk = idx & 1;
      if (idx < 15) {
        const int tn = (idx + 1) >> 1;
        const int kn = (idx + 1) & 1;
        const unsigned short* brow = &Bs[c][tn * 16 + l16][kn * 32];
        bnxt.q[0] = *(const u32x4*)(brow + kbb);
        bnxt.q[1] = *(const u32x4*)(brow + kbb + 16);
      }
      acc[t] = __builtin_amdgcn_wmma_f32_16x16x32_bf16(
          false, (kk ? af1.bf : af0.bf), false, bcur.bf, (short)0, acc[t],
          false, false);
      bcur = bnxt;
    }

    if (!last) {
      if (waveS == 0) __builtin_amdgcn_s_wait_tensorcnt(0);
      __syncthreads();
      c ^= 1;
    }
  }
#else
  // Fallback: cooperative global->LDS staging, K-step 64.
  const int ldRow = tid >> 1;
  const int ldCol = (tid & 1) * 32;
  for (int k0 = 0; k0 < K; k0 += 64) {
    const unsigned short* aSrc = A + (size_t)(bm + ldRow) * K + k0 + ldCol;
    const unsigned short* wSrc = W + (size_t)(bn + ldRow) * K + k0 + ldCol;
    u32x4 av[4], wv[4];
#pragma unroll
    for (int i = 0; i < 4; ++i) {
      av[i] = *(const u32x4*)(aSrc + i * 8);
      wv[i] = *(const u32x4*)(wSrc + i * 8);
    }
    __syncthreads();
#pragma unroll
    for (int i = 0; i < 4; ++i) {
      *(u32x4*)&As[0][ldRow][ldCol + i * 8] = av[i];
      *(u32x4*)&Bs[0][ldRow][ldCol + i * 8] = wv[i];
    }
    __syncthreads();
    const unsigned short* arow = &As[0][waveS * 16 + l16][0];
    FragBF af0, af1;
    af0.q[0] = *(const u32x4*)(arow + ka);
    af0.q[1] = *(const u32x4*)(arow + 16 + ka);
    af1.q[0] = *(const u32x4*)(arow + 32 + ka);
    af1.q[1] = *(const u32x4*)(arow + 48 + ka);
#pragma unroll
    for (int idx = 0; idx < 16; ++idx) {
      const int t = idx >> 1;
      const int kk = idx & 1;
      FragBF bfrag;
      const unsigned short* brow = &Bs[0][t * 16 + l16][kk * 32];
      bfrag.q[0] = *(const u32x4*)(brow + kbb);
      bfrag.q[1] = *(const u32x4*)(brow + kbb + 16);
      acc[t] = __builtin_amdgcn_wmma_f32_16x16x32_bf16(
          false, (kk ? af1.bf : af0.bf), false, bfrag.bf, (short)0, acc[t],
          false, false);
    }
  }
#endif

  // Epilogue: C layout lane L -> (M = r + lhalf*8, N = l16) per VGPR r
#pragma unroll
  for (int t = 0; t < 8; ++t) {
    const int gn = bn + t * 16 + l16;
    const float bv = bias ? bias[gn] : 0.0f;
#pragma unroll
    for (int r = 0; r < 8; ++r) {
      const int gm = bm + waveS * 16 + r + lhalf * 8;
      C[(size_t)gm * N + gn] = acc[t][r] + bv;
    }
  }
}

// ---------------------------------------------------------------------------
// Kernel 3: xPos rotary, fold in HEAD_DIM^-0.5, pack bf16.
// ---------------------------------------------------------------------------
__global__ __launch_bounds__(128) void rope_xpos_pack(
    const float* __restrict__ qf, const float* __restrict__ kvf,
    unsigned short* __restrict__ qb, unsigned short* __restrict__ kb,
    unsigned short* __restrict__ vb) {
  const int bs = blockIdx.x;
  const int b = bs / SEQ;
  const int s = bs - b * SEQ;
  const int d = threadIdx.x;
  const int j = d & 63;

  const float inv_freq =
      __expf(-(float)(2 * j) * (1.0f / (float)HEAD_DIM) * 9.210340372f);
  const float pos = (float)s * inv_freq;
  float sn, cs;
  __sincosf(pos, &sn, &cs);
  const float sbase =
      (2.0f * (float)j + 0.4f * (float)HEAD_DIM) / (1.4f * (float)HEAD_DIM);
  const float power = ((float)s - (float)(SEQ / 2)) * (1.0f / 512.0f);
  const float xs = __expf(power * __logf(sbase));
  const float inv_xs = 1.0f / xs;
  const float scale_w = 0.0883883476483f; // 128^-0.5

  const float* kvrow = kvf + (size_t)bs * (2 * HEAD_DIM);
  {
    const float kd = kvrow[d] * scale_w;
    const float ko = (d < 64 ? -kvrow[d + 64] : kvrow[d - 64]) * scale_w;
    kb[(size_t)bs * HEAD_DIM + d] = f2bf((kd * cs + ko * sn) * inv_xs);
    vb[(size_t)bs * HEAD_DIM + d] = f2bf(kvrow[HEAD_DIM + d]);
  }
  const float* qrow = qf + (size_t)bs * D_MODEL;
#pragma unroll 4
  for (int h = 0; h < N_HEADS; ++h) {
    const float qd = qrow[h * HEAD_DIM + d] * scale_w;
    const float qo =
        (d < 64 ? -qrow[h * HEAD_DIM + d + 64] : qrow[h * HEAD_DIM + d - 64]) *
        scale_w;
    qb[(((size_t)(b * N_HEADS + h)) * SEQ + s) * HEAD_DIM + d] =
        f2bf((qd * cs + qo * sn) * xs);
  }
}

// ---------------------------------------------------------------------------
// Kernel 4: causal flash attention. Block = 128 query rows x one (b,h),
// 8 waves, wave-private 16-row softmax state, 32-key tiles.
// K-tile arrives via TDM (wave 0), V transposed by all threads in parallel.
// ---------------------------------------------------------------------------
__global__ __launch_bounds__(256) void flash_attn_wmma(
    const unsigned short* __restrict__ Q,  // [B*H, S, Dh]
    const unsigned short* __restrict__ Kk, // [B, S, Dh]
    const unsigned short* __restrict__ V,  // [B, S, Dh]
    unsigned short* __restrict__ O) {      // [B, S, H*Dh]
  // K row = 128 elems = 64 DW -> pad_interval code 5; pad 4 DW (code 3)
  // -> row stride 272 B == [136] ushorts.
  __shared__ unsigned short Ks[32][136];   // [key][d]
  __shared__ unsigned short Vt[128][40];   // transposed [d][key], +8 pad
  __shared__ unsigned short Pb[8][16][40]; // per-wave P bounce

  const int tid = threadIdx.x;
  const int waveS = __builtin_amdgcn_readfirstlane(tid) >> 5;
  const int lane = tid & 31;
  const int lhalf = lane >> 4;
  const int l16 = lane & 15;

  const int rowblk = blockIdx.x;
  const int bh = blockIdx.y;
  const int b = bh >> 4;
  const int h = bh & 15;

  const int qrow0 = rowblk * 128 + waveS * 16;

  FragBF qa[4];
  {
    const unsigned short* qp =
        Q + ((size_t)bh * SEQ + qrow0 + l16) * HEAD_DIM + lhalf * 8;
#pragma unroll
    for (int kk = 0; kk < 4; ++kk) {
      qa[kk].q[0] = *(const u32x4*)(qp + kk * 32);
      qa[kk].q[1] = *(const u32x4*)(qp + kk * 32 + 16);
    }
  }

  v8f Oacc[8];
#pragma unroll
  for (int t = 0; t < 8; ++t)
#pragma unroll
    for (int r = 0; r < 8; ++r) Oacc[t][r] = 0.0f;
  float mi[8], li[8];
#pragma unroll
  for (int r = 0; r < 8; ++r) {
    mi[r] = -3.0e38f;
    li[r] = 0.0f;
  }

  const int njb = (rowblk + 1) * 4;
  const int ldRow = tid >> 3;       // 0..31 (key)
  const int ldCol = (tid & 7) * 16; // 0..112 (d)
  const int kb = lhalf * 8;

  for (int jb = 0; jb < njb; ++jb) {
    const int key0 = jb * 32;
    const unsigned short* vSrc =
        V + ((size_t)b * SEQ + key0 + ldRow) * HEAD_DIM + ldCol;
    u32x4 vd0 = *(const u32x4*)(vSrc);
    u32x4 vd1 = *(const u32x4*)(vSrc + 8);

    __syncthreads(); // everyone done reading Ks/Vt from previous tile
#if HAVE_TDM
    if (waveS == 0)
      tdm_load_2d((unsigned)(size_t)&Ks[0][0],
                  Kk + ((size_t)b * SEQ + key0) * HEAD_DIM, 128, 32,
                  (unsigned)HEAD_DIM, (unsigned)SEQ, (unsigned)HEAD_DIM, 5, 3);
#else
    {
      const unsigned short* kSrc =
          Kk + ((size_t)b * SEQ + key0 + ldRow) * HEAD_DIM + ldCol;
      u32x4 kd0 = *(const u32x4*)(kSrc);
      u32x4 kd1 = *(const u32x4*)(kSrc + 8);
      *(u32x4*)&Ks[ldRow][ldCol] = kd0;
      *(u32x4*)&Ks[ldRow][ldCol + 8] = kd1;
    }
#endif
    { // V transpose-on-store (overlaps the TDM fill)
      const unsigned short* vv = (const unsigned short*)&vd0;
#pragma unroll
      for (int i = 0; i < 8; ++i) Vt[ldCol + i][ldRow] = vv[i];
      vv = (const unsigned short*)&vd1;
#pragma unroll
      for (int i = 0; i < 8; ++i) Vt[ldCol + 8 + i][ldRow] = vv[i];
    }
#if HAVE_TDM
    if (waveS == 0) __builtin_amdgcn_s_wait_tensorcnt(0);
#endif
    __syncthreads();

    // scores: S(16x32) = Q(16x128) @ K^T, 8 WMMAs
    v8f sc[2];
#pragma unroll
    for (int t = 0; t < 2; ++t)
#pragma unroll
      for (int r = 0; r < 8; ++r) sc[t][r] = 0.0f;
#pragma unroll
    for (int t = 0; t < 2; ++t) {
      const unsigned short* krow = &Ks[t * 16 + l16][0];
#pragma unroll
      for (int kk = 0; kk < 4; ++kk) {
        FragBF bfrag;
        bfrag.q[0] = *(const u32x4*)(krow + kk * 32 + kb);
        bfrag.q[1] = *(const u32x4*)(krow + kk * 32 + kb + 16);
        sc[t] = __builtin_amdgcn_wmma_f32_16x16x32_bf16(
            false, qa[kk].bf, false, bfrag.bf, (short)0, sc[t], false, false);
      }
    }

    // causal mask
#pragma unroll
    for (int t = 0; t < 2; ++t) {
      const int kcol = key0 + t * 16 + l16;
#pragma unroll
      for (int r = 0; r < 8; ++r) {
        const int qr = qrow0 + r + lhalf * 8;
        if (kcol > qr) sc[t][r] = -3.0e38f;
      }
    }

    // online softmax (16-lane reductions stay within half-waves)
#pragma unroll
    for (int r = 0; r < 8; ++r) {
      float rm = fmaxf(sc[0][r], sc[1][r]);
#pragma unroll
      for (int off = 8; off >= 1; off >>= 1)
        rm = fmaxf(rm, __shfl_xor(rm, off, 32));
      const float mnew = fmaxf(mi[r], rm);
      const float corr = __expf(mi[r] - mnew);
      mi[r] = mnew;
      const float p0 = __expf(sc[0][r] - mnew);
      const float p1 = __expf(sc[1][r] - mnew);
      sc[0][r] = p0;
      sc[1][r] = p1;
      float rs = p0 + p1;
#pragma unroll
      for (int off = 8; off >= 1; off >>= 1) rs += __shfl_xor(rs, off, 32);
      li[r] = li[r] * corr + rs;
#pragma unroll
      for (int t = 0; t < 8; ++t) Oacc[t][r] *= corr;
    }

    // P: C-layout -> bf16 A-layout via wave-private LDS bounce
#pragma unroll
    for (int t = 0; t < 2; ++t)
#pragma unroll
      for (int r = 0; r < 8; ++r)
        Pb[waveS][r + lhalf * 8][t * 16 + l16] = f2bf(sc[t][r]);
    FragBF pa;
    pa.q[0] = *(const u32x4*)&Pb[waveS][l16][lhalf * 8];
    pa.q[1] = *(const u32x4*)&Pb[waveS][l16][16 + lhalf * 8];

    // O += P @ V, 8 WMMAs
    FragBF vf[8];
#pragma unroll
    for (int t = 0; t < 8; ++t) {
      vf[t].q[0] = *(const u32x4*)&Vt[t * 16 + l16][kb];
      vf[t].q[1] = *(const u32x4*)&Vt[t * 16 + l16][kb + 16];
    }
#pragma unroll
    for (int t = 0; t < 8; ++t)
      Oacc[t] = __builtin_amdgcn_wmma_f32_16x16x32_bf16(
          false, pa.bf, false, vf[t].bf, (short)0, Oacc[t], false, false);
  }

  // epilogue: O /= l, write bf16 context [B,S,H*Dh]
  unsigned short* oBase = O + (size_t)b * SEQ * D_MODEL + (size_t)h * HEAD_DIM;
#pragma unroll
  for (int r = 0; r < 8; ++r) {
    const int srow = qrow0 + r + lhalf * 8;
    const float inv_l = 1.0f / li[r];
#pragma unroll
    for (int t = 0; t < 8; ++t)
      oBase[(size_t)srow * D_MODEL + t * 16 + l16] = f2bf(Oacc[t][r] * inv_l);
  }
}

// ---------------------------------------------------------------------------
// Host-side launcher
// ---------------------------------------------------------------------------
extern "C" void kernel_launch(void* const* d_in, const int* in_sizes, int n_in,
                              void* d_out, int out_size, void* d_ws,
                              size_t ws_size, hipStream_t stream) {
  (void)in_sizes; (void)n_in; (void)out_size; (void)ws_size;
  const float* hs     = (const float*)d_in[0];
  const float* w_q    = (const float*)d_in[1];
  const float* w_kv   = (const float*)d_in[2];
  const float* w_proj = (const float*)d_in[3];
  const float* b_proj = (const float*)d_in[4];

  char* ws = (char*)d_ws;
  unsigned short* hs_bf  = (unsigned short*)(ws + 0);
  unsigned short* wq_bf  = (unsigned short*)(ws + 16777216);
  unsigned short* wkv_bf = (unsigned short*)(ws + 25165824);
  unsigned short* wp_bf  = (unsigned short*)(ws + 26214400);
  float*          q_f32  = (float*)(ws + 34603008);
  float*          kv_f32 = (float*)(ws + 68157440);
  unsigned short* q_bf   = (unsigned short*)(ws + 72351744);
  unsigned short* k_bf   = (unsigned short*)(ws + 89128960);
  unsigned short* v_bf   = (unsigned short*)(ws + 90177536);
  unsigned short* ctx_bf = (unsigned short*)(ws + 34603008); // aliases q_f32

  cvt_f32_bf16<<<dim3(8192), dim3(256), 0, stream>>>(hs, hs_bf, MTOT * D_MODEL);
  cvt_f32_bf16<<<dim3(4096), dim3(256), 0, stream>>>(w_q, wq_bf, D_MODEL * D_MODEL);
  cvt_f32_bf16<<<dim3(512), dim3(256), 0, stream>>>(w_kv, wkv_bf, 2 * HEAD_DIM * D_MODEL);
  cvt_f32_bf16<<<dim3(4096), dim3(256), 0, stream>>>(w_proj, wp_bf, D_MODEL * D_MODEL);

  gemm_bf16_wmma<<<dim3(32, 16), 256, 0, stream>>>(hs_bf, wq_bf, nullptr, q_f32,
                                                   MTOT, D_MODEL, D_MODEL);
  gemm_bf16_wmma<<<dim3(32, 2), 256, 0, stream>>>(hs_bf, wkv_bf, nullptr,
                                                  kv_f32, MTOT, 2 * HEAD_DIM,
                                                  D_MODEL);

  rope_xpos_pack<<<dim3(MTOT), 128, 0, stream>>>(q_f32, kv_f32, q_bf, k_bf, v_bf);

  flash_attn_wmma<<<dim3(SEQ / 128, BATCH * N_HEADS), 256, 0, stream>>>(
      q_bf, k_bf, v_bf, ctx_bf);

  gemm_bf16_wmma<<<dim3(32, 16), 256, 0, stream>>>(ctx_bf, wp_bf, b_proj,
                                                   (float*)d_out, MTOT, D_MODEL,
                                                   D_MODEL);
}
